// LightweightTransformer_9603546873925
// MI455X (gfx1250) — matrix operand order repaired and checked by reading, not compiled, not run on hardware
//
#include <hip/hip_runtime.h>
#include <hip/hip_bf16.h>
#include <math.h>

// ---------------------------------------------------------------------------
// Problem constants (from the reference)
// ---------------------------------------------------------------------------
#define BB   16
#define CIN  2
#define LL   8192
#define DD   256
#define HH   8
#define NLAY 4
#define DKK  32
#define DFF_ 512
#define SS   1024          // downsampled length
#define MM   (BB * SS)     // 16384 rows for all token GEMMs
#define L1   2048          // conv1 output length

typedef __bf16 bf16;
typedef __attribute__((ext_vector_type(16))) __bf16 v16bf;
typedef __attribute__((ext_vector_type(2)))  __bf16 bf16x2;
typedef __attribute__((ext_vector_type(8)))  float  v8f;

// ---------------------------------------------------------------------------
// Helpers
// ---------------------------------------------------------------------------
__device__ __forceinline__ bf16 f2bf(float f) {
  unsigned u = __builtin_bit_cast(unsigned, f);
  unsigned r = u + 0x7FFFu + ((u >> 16) & 1u);   // round-to-nearest-even
  unsigned short h = (unsigned short)(r >> 16);
  return __builtin_bit_cast(bf16, h);
}

__device__ __forceinline__ v8f wmma_bf16(v16bf a, v16bf b, v8f c) {
  // (neg_a, A, neg_b, B, c_mod, C, reuse_a, reuse_b)
  return __builtin_amdgcn_wmma_f32_16x16x32_bf16(false, a, false, b,
                                                 (short)0, c, false, false);
}

// CDNA5 async copy: global -> LDS, 16 bytes per lane, tracked by ASYNCcnt.
// LDS address = low 32 bits of the generic shared pointer (wave-relative).
__device__ __forceinline__ void async_copy_b128(void* lds_dst, const void* gsrc) {
  unsigned loff = (unsigned)(unsigned long long)lds_dst;
  asm volatile("global_load_async_to_lds_b128 %0, %1, off"
               :: "v"(loff), "v"(gsrc) : "memory");
}
__device__ __forceinline__ void wait_async_0() {
  asm volatile("s_wait_asynccnt 0" ::: "memory");
}
__device__ __forceinline__ void wait_async_4() {
  asm volatile("s_wait_asynccnt 4" ::: "memory");
}

// A-fragment: 16x32 (MxK), row-major A, k = 16*(lane/16) + element (contiguous order, matched by the B loaders).
// lane: row = lane&15 ; lanes>=16 offset K by +16 ; VGPR j -> K=2j
__device__ __forceinline__ v16bf load_a_frag(const bf16* A, int lda) {
  int lane = threadIdx.x & 31;
  int row  = lane & 15;
  int koff = (lane >> 4) * 16;
  const bf16* r = A + (size_t)row * lda + koff;
  v16bf a;
#pragma unroll
  for (int j = 0; j < 8; ++j) {
    int k = 2 * j;
    bf16x2 p = *reinterpret_cast<const bf16x2*>(r + k);
    a[2 * j]     = p[0];
    a[2 * j + 1] = p[1];
  }
  return a;
}

// B-fragment: 32x16 (KxN), operand stored N-major with K contiguous.
__device__ __forceinline__ v16bf load_b_frag_kcont(const bf16* Bt, int ldb) {
  int lane = threadIdx.x & 31;
  int n    = lane & 15;
  int kb   = (lane >> 4) * 16;
  const bf16* r = Bt + (size_t)n * ldb + kb;
  v16bf b;
#pragma unroll
  for (int j = 0; j < 8; ++j) {
    bf16x2 p = *reinterpret_cast<const bf16x2*>(r + 2 * j);
    b[2 * j]     = p[0];
    b[2 * j + 1] = p[1];
  }
  return b;
}

// B-fragment from row-major KxN storage (K rows are ldb elements apart).
__device__ __forceinline__ v16bf load_b_frag_krows(const bf16* Bp, int ldb) {
  int lane = threadIdx.x & 31;
  int n    = lane & 15;
  int kb   = (lane >> 4) * 16;
  v16bf b;
#pragma unroll
  for (int j = 0; j < 8; ++j) {
    b[2 * j]     = Bp[(size_t)(kb + 2 * j)     * ldb + n];
    b[2 * j + 1] = Bp[(size_t)(kb + 2 * j + 1) * ldb + n];
  }
  return b;
}

// ---------------------------------------------------------------------------
// Weight convert + transpose: src f32 (K,N) -> dst bf16 (N,K)
// ---------------------------------------------------------------------------
__global__ void cvt_transpose_kernel(const float* __restrict__ src,
                                     bf16* __restrict__ dst, int K, int N) {
  int idx = blockIdx.x * blockDim.x + threadIdx.x;
  if (idx >= K * N) return;
  int n = idx % N, k = idx / N;
  dst[(size_t)n * K + k] = f2bf(src[idx]);
}

// ---------------------------------------------------------------------------
// Conv1: (B,2,8192) -> relu -> (B,32,2048), k=8 s=4 p=2
// ---------------------------------------------------------------------------
__global__ void conv1_kernel(const float* __restrict__ x,
                             const float* __restrict__ w,
                             const float* __restrict__ bias,
                             float* __restrict__ out) {
  int idx = blockIdx.x * blockDim.x + threadIdx.x;
  if (idx >= BB * 32 * L1) return;
  int t  = idx & (L1 - 1);
  int oc = (idx >> 11) & 31;
  int b  = idx >> 16;
  float acc = bias[oc];
  const float* xb = x + (size_t)b * CIN * LL;
  const float* wr = w + (size_t)oc * CIN * 8;
#pragma unroll
  for (int ic = 0; ic < CIN; ++ic) {
#pragma unroll
    for (int j = 0; j < 8; ++j) {
      int p = 4 * t - 2 + j;
      if (p >= 0 && p < LL) acc += xb[ic * LL + p] * wr[ic * 8 + j];
    }
  }
  out[idx] = fmaxf(acc, 0.f);
}

// ---------------------------------------------------------------------------
// Conv2 + relu + transpose + positional encoding: -> h (B,S,D) f32 + bf16
// ---------------------------------------------------------------------------
__global__ void conv2_pe_kernel(const float* __restrict__ h1,
                                const float* __restrict__ w,
                                const float* __restrict__ bias,
                                float* __restrict__ hF,
                                bf16* __restrict__ hB) {
  int idx = blockIdx.x * blockDim.x + threadIdx.x;
  if (idx >= BB * SS * DD) return;
  int d = idx & (DD - 1);
  int s = (idx >> 8) & (SS - 1);
  int b = idx >> 18;
  float acc = bias[d];
  const float* hb = h1 + (size_t)b * 32 * L1;
  const float* wr = w + (size_t)d * 32 * 4;
#pragma unroll 4
  for (int ic = 0; ic < 32; ++ic) {
#pragma unroll
    for (int j = 0; j < 4; ++j) {
      int p = 2 * s - 1 + j;
      if (p >= 0 && p < L1) acc += hb[ic * L1 + p] * wr[ic * 4 + j];
    }
  }
  acc = fmaxf(acc, 0.f);
  int i2 = d >> 1;
  float div = __expf(-(float)(2 * i2) * (9.210340371976184f / (float)DD)); // ln(10000)
  float ang = (float)s * div;
  acc += (d & 1) ? __cosf(ang) : __sinf(ang);
  hF[idx] = acc;
  hB[idx] = f2bf(acc);
}

// ---------------------------------------------------------------------------
// WMMA GEMM: Out[M,N] = act(A_bf16[M,K] @ W + bias) ; W stored (N,K) bf16.
// Block = 256 threads = 8 waves; block tile 128x64, wave tile 16x64.
// Weight tile (64x32) double-buffered in LDS via GLOBAL_LOAD_ASYNC_TO_LDS,
// pipelined one K-step ahead (ASYNCcnt).  Row pitch 56 -> bank-conflict-free
// 16-row fragment reads.
// ---------------------------------------------------------------------------
#define WPITCH 56
#define WTELEM (64 * WPITCH)

__global__ __launch_bounds__(256) void gemm_wmma_kernel(
    const bf16* __restrict__ A, const bf16* __restrict__ Wt,
    const float* __restrict__ bias, float* __restrict__ outF,
    bf16* __restrict__ outB, int M, int N, int K, int relu) {
  __shared__ bf16 wtile[2 * WTELEM];

  int wave = threadIdx.x >> 5;
  int m0 = blockIdx.y * 128 + wave * 16;
  int n0 = blockIdx.x * 64;

  // cooperative staging coordinates: 64 rows x 4 segments of 16B
  int sn = threadIdx.x >> 2;
  int sg = threadIdx.x & 3;
  const bf16* wsrc = Wt + (size_t)(n0 + sn) * K + sg * 8;
  bf16* wdst = wtile + sn * WPITCH + sg * 8;

  async_copy_b128(wdst, wsrc);                       // stage tile for k0 = 0

  v8f acc0 = {}, acc1 = {}, acc2 = {}, acc3 = {};
  const bf16* Arow = A + (size_t)m0 * K;

  for (int k0 = 0; k0 < K; k0 += 32) {
    int buf = (k0 >> 5) & 1;
    wait_async_0();          // this wave's share of tile k0 has landed
    __syncthreads();         // whole tile k0 visible; prev reads of buf^1 done
    if (k0 + 32 < K)         // stream next tile behind the WMMAs
      async_copy_b128(wdst + (buf ^ 1) * WTELEM, wsrc + (k0 + 32));

    if (k0 + 32 < K) __builtin_prefetch(Arow + k0 + 32, 0, 1);
    v16bf a = load_a_frag(Arow + k0, K);
    const bf16* wb = wtile + buf * WTELEM;
    v16bf b0 = load_b_frag_kcont(wb,               WPITCH);
    v16bf b1 = load_b_frag_kcont(wb + 16 * WPITCH, WPITCH);
    v16bf b2 = load_b_frag_kcont(wb + 32 * WPITCH, WPITCH);
    v16bf b3 = load_b_frag_kcont(wb + 48 * WPITCH, WPITCH);
    acc0 = wmma_bf16(a, b0, acc0);
    acc1 = wmma_bf16(a, b1, acc1);
    acc2 = wmma_bf16(a, b2, acc2);
    acc3 = wmma_bf16(a, b3, acc3);
  }

  int lane = threadIdx.x & 31;
  int nn = lane & 15;
  int mb = (lane >> 4) * 8;
  v8f accs[4] = {acc0, acc1, acc2, acc3};
#pragma unroll
  for (int t = 0; t < 4; ++t) {
    int col = n0 + t * 16 + nn;
    float bv = bias ? bias[col] : 0.f;
#pragma unroll
    for (int v = 0; v < 8; ++v) {
      float c = accs[t][v] + bv;
      if (relu) c = fmaxf(c, 0.f);
      size_t o = (size_t)(m0 + mb + v) * N + col;
      if (outF) outF[o] = c;
      if (outB) outB[o] = f2bf(c);
    }
  }
}

// ---------------------------------------------------------------------------
// Attention: one workgroup (128 thr = 4 waves) per (query-block-16, head, batch)
// scores 16x1024 in LDS -> softmax -> bf16 probs -> P@V via WMMA.
// V chunks (32 keys x 32 dk) are double-buffered per wave in LDS via async
// copies (4x b128 per lane, pitch 40 -> 16B aligned, conflict-free).
// ---------------------------------------------------------------------------
#define VPITCH 40
#define VSELEM (32 * VPITCH)

__global__ __launch_bounds__(128) void attn_kernel(
    const bf16* __restrict__ Q, const bf16* __restrict__ Kb,
    const bf16* __restrict__ Vb, bf16* __restrict__ ctxB) {
  extern __shared__ char smem_raw[];
  float* sc    = (float*)smem_raw;                                   // 16*1024 f32
  bf16*  pb    = (bf16*)(smem_raw + 16 * 1024 * 4);                  // 16*1024 bf16
  float* part  = (float*)(smem_raw + 16 * 1024 * 4 + 16 * 1024 * 2); // 4*2*16*16
  float* red   = part + 4 * 2 * 16 * 16;                             // 16*8
  float* rstat = red + 16 * 8;                                       // 32
  bf16*  vsall = (bf16*)(rstat + 32);                                // 4*2*VSELEM

  int qb = blockIdx.x, hd = blockIdx.y, b = blockIdx.z;
  int wave = threadIdx.x >> 5;
  int lane = threadIdx.x & 31;
  int nn = lane & 15, mb = (lane >> 4) * 8;

  const float scale = 0.17677669529663687f;  // 1/sqrt(32)

  // ---- scores = Q @ K^T ----
  const bf16* Qp = Q + ((size_t)(b * SS + qb * 16) * DD + hd * DKK);
  v16bf qf = load_a_frag(Qp, DD);
  for (int kt = wave; kt < SS / 16; kt += 4) {
    const bf16* Kp = Kb + ((size_t)(b * SS + kt * 16) * DD + hd * DKK);
    v16bf kf = load_b_frag_kcont(Kp, DD);
    v8f c = {};
    c = wmma_bf16(qf, kf, c);
#pragma unroll
    for (int v = 0; v < 8; ++v)
      sc[(mb + v) * SS + kt * 16 + nn] = c[v] * scale;
  }

  // ---- start streaming this wave's first two V chunks (hidden by softmax) --
  bf16* vst = vsall + wave * 2 * VSELEM;
  int cbase = wave * 8;  // 8 chunks of 32 keys per wave
  {
    const bf16* Vp0 = Vb + ((size_t)(b * SS + cbase * 32) * DD + hd * DKK);
    const bf16* Vp1 = Vb + ((size_t)(b * SS + (cbase + 1) * 32) * DD + hd * DKK);
#pragma unroll
    for (int s2 = 0; s2 < 4; ++s2)
      async_copy_b128(vst + lane * VPITCH + s2 * 8,
                      Vp0 + (size_t)lane * DD + s2 * 8);
#pragma unroll
    for (int s2 = 0; s2 < 4; ++s2)
      async_copy_b128(vst + VSELEM + lane * VPITCH + s2 * 8,
                      Vp1 + (size_t)lane * DD + s2 * 8);
  }
  __syncthreads();

  // ---- softmax over 16 rows of length 1024 (8 threads per row) ----
  int tid = threadIdx.x;
  int r = tid >> 3, sub = tid & 7;
  float mx = -1e30f;
  for (int c2 = sub * 128; c2 < sub * 128 + 128; ++c2)
    mx = fmaxf(mx, sc[r * SS + c2]);
  red[r * 8 + sub] = mx;
  __syncthreads();
  if (sub == 0) {
    float m2 = red[r * 8];
    for (int j = 1; j < 8; ++j) m2 = fmaxf(m2, red[r * 8 + j]);
    rstat[r] = m2;
  }
  __syncthreads();
  float rm = rstat[r];
  float sum = 0.f;
  for (int c2 = sub * 128; c2 < sub * 128 + 128; ++c2) {
    float e = __expf(sc[r * SS + c2] - rm);
    sc[r * SS + c2] = e;
    sum += e;
  }
  red[r * 8 + sub] = sum;
  __syncthreads();
  if (sub == 0) {
    float s2 = 0.f;
    for (int j = 0; j < 8; ++j) s2 += red[r * 8 + j];
    rstat[16 + r] = 1.f / s2;
  }
  __syncthreads();
  float inv = rstat[16 + r];
  for (int c2 = sub * 128; c2 < sub * 128 + 128; ++c2)
    pb[r * SS + c2] = f2bf(sc[r * SS + c2] * inv);
  __syncthreads();

  // ---- ctx = P @ V, pipelined over this wave's 8 key chunks ----
  v8f ca0 = {}, ca1 = {};
  for (int ci = 0; ci < 8; ++ci) {
    int buf = ci & 1;
    // in-order ASYNCcnt: <=4 outstanding means chunk ci has landed
    if (ci < 6) wait_async_4(); else wait_async_0();

    v16bf pf = load_a_frag((const bf16*)pb + (cbase + ci) * 32, SS);
    const bf16* vbuf = vst + buf * VSELEM;
    v16bf v0 = load_b_frag_krows(vbuf,      VPITCH);
    v16bf v1 = load_b_frag_krows(vbuf + 16, VPITCH);
    ca0 = wmma_bf16(pf, v0, ca0);
    ca1 = wmma_bf16(pf, v1, ca1);

    if (ci + 2 < 8) {   // refill this buffer (its reads are consumed above)
      const bf16* Vpn = Vb + ((size_t)(b * SS + (cbase + ci + 2) * 32) * DD + hd * DKK);
#pragma unroll
      for (int s2 = 0; s2 < 4; ++s2)
        async_copy_b128(vst + buf * VSELEM + lane * VPITCH + s2 * 8,
                        Vpn + (size_t)lane * DD + s2 * 8);
    }
  }
#pragma unroll
  for (int v = 0; v < 8; ++v) {
    part[((wave * 2 + 0) * 16 + (mb + v)) * 16 + nn] = ca0[v];
    part[((wave * 2 + 1) * 16 + (mb + v)) * 16 + nn] = ca1[v];
  }
  __syncthreads();

  // cross-wave reduce of 2 tiles x 256 elements
  for (int e = tid; e < 512; e += 128) {
    int t = e >> 8, idx = e & 255, m = idx >> 4, n2 = idx & 15;
    float s = 0.f;
    for (int w2 = 0; w2 < 4; ++w2) s += part[((w2 * 2 + t) * 16 + m) * 16 + n2];
    ctxB[((size_t)(b * SS + qb * 16 + m)) * DD + hd * DKK + t * 16 + n2] = f2bf(s);
  }
}

// ---------------------------------------------------------------------------
// Residual + LayerNorm (in-place on h), also emits bf16 copy. 1 block = 1 row.
// ---------------------------------------------------------------------------
__global__ __launch_bounds__(256) void resid_ln_kernel(
    float* __restrict__ h, const float* __restrict__ delta,
    const float* __restrict__ g, const float* __restrict__ bta,
    bf16* __restrict__ hB) {
  __shared__ float redS[256];
  int row = blockIdx.x, d = threadIdx.x;
  size_t o = (size_t)row * DD + d;
  float x = h[o] + delta[o];
  redS[d] = x;
  __syncthreads();
  for (int s = 128; s > 0; s >>= 1) {
    if (d < s) redS[d] += redS[d + s];
    __syncthreads();
  }
  float mean = redS[0] * (1.f / DD);
  __syncthreads();
  float c = x - mean;
  redS[d] = c * c;
  __syncthreads();
  for (int s = 128; s > 0; s >>= 1) {
    if (d < s) redS[d] += redS[d + s];
    __syncthreads();
  }
  float var = redS[0] * (1.f / DD);
  float y = c * rsqrtf(var + 1e-5f) * g[d] + bta[d];
  h[o] = y;
  hB[o] = f2bf(y);
}

// ---------------------------------------------------------------------------
// Mean pool + classifier head + sigmoid. 1 block per batch element.
// ---------------------------------------------------------------------------
__global__ __launch_bounds__(256) void cls_kernel(
    const float* __restrict__ h, const float* __restrict__ w1,
    const float* __restrict__ b1, const float* __restrict__ w2,
    const float* __restrict__ b2, float* __restrict__ out) {
  __shared__ float pooled[DD];
  __shared__ float hid[DD / 2];
  int b = blockIdx.x, d = threadIdx.x;
  const float* hb = h + (size_t)b * SS * DD;
  float s = 0.f;
  for (int t = 0; t < SS; ++t) s += hb[(size_t)t * DD + d];
  pooled[d] = s * (1.f / SS);
  __syncthreads();
  if (d < DD / 2) {
    float a = b1[d];
    for (int k = 0; k < DD; ++k) a += pooled[k] * w1[k * (DD / 2) + d];
    hid[d] = fmaxf(a, 0.f);
  }
  __syncthreads();
  if (d == 0) {
    float a = b2[0];
    for (int k = 0; k < DD / 2; ++k) a += hid[k] * w2[k];
    out[b] = 1.f / (1.f + __expf(-a));
  }
}

// ---------------------------------------------------------------------------
// Host orchestration
// ---------------------------------------------------------------------------
extern "C" void kernel_launch(void* const* d_in, const int* in_sizes, int n_in,
                              void* d_out, int out_size, void* d_ws, size_t ws_size,
                              hipStream_t stream) {
  const float* x       = (const float*)d_in[0];
  const float* conv1_w = (const float*)d_in[1];
  const float* conv1_b = (const float*)d_in[2];
  const float* conv2_w = (const float*)d_in[3];
  const float* conv2_b = (const float*)d_in[4];
  const float* wq = (const float*)d_in[5];
  const float* bq = (const float*)d_in[6];
  const float* wk = (const float*)d_in[7];
  const float* bk = (const float*)d_in[8];
  const float* wv = (const float*)d_in[9];
  const float* bv = (const float*)d_in[10];
  const float* wo = (const float*)d_in[11];
  const float* bo = (const float*)d_in[12];
  const float* ln1_g = (const float*)d_in[13];
  const float* ln1_b = (const float*)d_in[14];
  const float* ff_w1 = (const float*)d_in[15];
  const float* ff_b1 = (const float*)d_in[16];
  const float* ff_w2 = (const float*)d_in[17];
  const float* ff_b2 = (const float*)d_in[18];
  const float* ln2_g = (const float*)d_in[19];
  const float* ln2_b = (const float*)d_in[20];
  const float* cls_w1 = (const float*)d_in[21];
  const float* cls_b1 = (const float*)d_in[22];
  const float* cls_w2 = (const float*)d_in[23];
  const float* cls_b2 = (const float*)d_in[24];
  float* out = (float*)d_out;

  // workspace carve-out
  size_t off = 0;
  auto alloc = [&](size_t bytes) -> char* {
    char* p = (char*)d_ws + off;
    off += (bytes + 255) & ~(size_t)255;
    return p;
  };
  bf16* wqt  = (bf16*)alloc((size_t)NLAY * DD * DD * 2);
  bf16* wkt  = (bf16*)alloc((size_t)NLAY * DD * DD * 2);
  bf16* wvt  = (bf16*)alloc((size_t)NLAY * DD * DD * 2);
  bf16* wot  = (bf16*)alloc((size_t)NLAY * DD * DD * 2);
  bf16* fw1t = (bf16*)alloc((size_t)NLAY * DD * DFF_ * 2);
  bf16* fw2t = (bf16*)alloc((size_t)NLAY * DFF_ * DD * 2);
  float* h1  = (float*)alloc((size_t)BB * 32 * L1 * 4);
  float* hF  = (float*)alloc((size_t)MM * DD * 4);
  bf16*  hB  = (bf16*) alloc((size_t)MM * DD * 2);
  bf16*  qb_ = (bf16*) alloc((size_t)MM * DD * 2);
  bf16*  kb_ = (bf16*) alloc((size_t)MM * DD * 2);
  bf16*  vb_ = (bf16*) alloc((size_t)MM * DD * 2);
  bf16*  ctxb = (bf16*)alloc((size_t)MM * DD * 2);
  float* projF = (float*)alloc((size_t)MM * DD * 4);
  bf16*  ff1b = (bf16*)alloc((size_t)MM * DFF_ * 2);
  (void)ws_size; (void)n_in; (void)in_sizes; (void)out_size;

  // ---- weight conversion (transpose to N-major, K-contiguous bf16) ----
  {
    int g_dd = (DD * DD + 255) / 256;
    int g_df = (DD * DFF_ + 255) / 256;
    for (int i = 0; i < NLAY; ++i) {
      cvt_transpose_kernel<<<g_dd, 256, 0, stream>>>(wq + (size_t)i * DD * DD, wqt + (size_t)i * DD * DD, DD, DD);
      cvt_transpose_kernel<<<g_dd, 256, 0, stream>>>(wk + (size_t)i * DD * DD, wkt + (size_t)i * DD * DD, DD, DD);
      cvt_transpose_kernel<<<g_dd, 256, 0, stream>>>(wv + (size_t)i * DD * DD, wvt + (size_t)i * DD * DD, DD, DD);
      cvt_transpose_kernel<<<g_dd, 256, 0, stream>>>(wo + (size_t)i * DD * DD, wot + (size_t)i * DD * DD, DD, DD);
      cvt_transpose_kernel<<<g_df, 256, 0, stream>>>(ff_w1 + (size_t)i * DD * DFF_, fw1t + (size_t)i * DD * DFF_, DD, DFF_);
      cvt_transpose_kernel<<<g_df, 256, 0, stream>>>(ff_w2 + (size_t)i * DFF_ * DD, fw2t + (size_t)i * DFF_ * DD, DFF_, DD);
    }
  }

  // ---- downsampler ----
  conv1_kernel<<<(BB * 32 * L1) / 256, 256, 0, stream>>>(x, conv1_w, conv1_b, h1);
  conv2_pe_kernel<<<(BB * SS * DD) / 256, 256, 0, stream>>>(h1, conv2_w, conv2_b, hF, hB);

  dim3 gemmDD(DD / 64, MM / 128);
  dim3 gemmDF(DFF_ / 64, MM / 128);
  dim3 attnG(SS / 16, HH, BB);
  size_t attn_smem = 16 * 1024 * 4        // scores f32
                   + 16 * 1024 * 2        // probs bf16
                   + 4 * 2 * 16 * 16 * 4  // partial ctx
                   + 16 * 8 * 4 + 32 * 4  // reductions
                   + 4 * 2 * VSELEM * 2;  // V staging (async)

  for (int i = 0; i < NLAY; ++i) {
    gemm_wmma_kernel<<<gemmDD, 256, 0, stream>>>(hB, wqt + (size_t)i * DD * DD, bq + i * DD,
                                                 nullptr, qb_, MM, DD, DD, 0);
    gemm_wmma_kernel<<<gemmDD, 256, 0, stream>>>(hB, wkt + (size_t)i * DD * DD, bk + i * DD,
                                                 nullptr, kb_, MM, DD, DD, 0);
    gemm_wmma_kernel<<<gemmDD, 256, 0, stream>>>(hB, wvt + (size_t)i * DD * DD, bv + i * DD,
                                                 nullptr, vb_, MM, DD, DD, 0);
    attn_kernel<<<attnG, 128, attn_smem, stream>>>(qb_, kb_, vb_, ctxb);
    gemm_wmma_kernel<<<gemmDD, 256, 0, stream>>>(ctxb, wot + (size_t)i * DD * DD, bo + i * DD,
                                                 projF, nullptr, MM, DD, DD, 0);
    resid_ln_kernel<<<MM, 256, 0, stream>>>(hF, projF, ln1_g + i * DD, ln1_b + i * DD, hB);
    gemm_wmma_kernel<<<gemmDF, 256, 0, stream>>>(hB, fw1t + (size_t)i * DD * DFF_, ff_b1 + i * DFF_,
                                                 nullptr, ff1b, MM, DFF_, DD, 1);
    gemm_wmma_kernel<<<gemmDD, 256, 0, stream>>>(ff1b, fw2t + (size_t)i * DFF_ * DD, ff_b2 + i * DD,
                                                 projF, nullptr, MM, DD, DFF_, 0);
    resid_ln_kernel<<<MM, 256, 0, stream>>>(hF, projF, ln2_g + i * DD, ln2_b + i * DD, hB);
  }

  cls_kernel<<<BB, 256, 0, stream>>>(hF, cls_w1, cls_b1, cls_w2, cls_b2, out);
}
